// DecoderBlock_80453327389080
// MI455X (gfx1250) — matrix operand order repaired and checked by reading, not compiled, not run on hardware
//
#include <hip/hip_runtime.h>

// ---------------------------------------------------------------------------
// MI455X (gfx1250) decoder block: bf16 WMMA GEMMs + flash attention.
// B=2, T=2048, C=768, H=12, dh=64.  ~79 GFLOP dense, ~100MB touched ->
// compute-bound on the WMMA pipes; fp32 only for LN / softmax / residuals.
// Uses gfx1250 async global->LDS copies for attention tile staging and
// global_prefetch in the GEMM K loop.
// ---------------------------------------------------------------------------

typedef __attribute__((ext_vector_type(16))) __bf16 v16bf;
typedef __attribute__((ext_vector_type(8)))  float  v8f;
typedef __attribute__((address_space(3)))    __bf16 lds_bf16_t;

#define LN_EPS 1e-3f

__device__ inline v8f wmma_bf16(v16bf a, v16bf b, v8f c) {
  // (neg_a, A, neg_b, B, c_mod, C, reuse_a, reuse_b)
  return __builtin_amdgcn_wmma_f32_16x16x32_bf16(false, a, false, b, (short)0, c,
                                                 false, false);
}

// 32-bit wave-relative LDS offset of a shared-memory pointer (for async ops:
// the LDS address is the VGPR value plus the instruction offset, so the VGPR wants the raw offset).
__device__ inline unsigned lds_off_of(__bf16* p) {
  return (unsigned)(size_t)(lds_bf16_t*)p;
}

// Async contiguous 16B global -> LDS copy (ASYNCcnt-tracked).
__device__ inline void async_copy_b128(unsigned lds_off, const __bf16* g) {
  asm volatile("global_load_async_to_lds_b128 %0, %1, off"
               :: "v"(lds_off), "v"((unsigned long long)(size_t)g)
               : "memory");
}

union AFrag { v16bf v; unsigned u[8]; };

// A operand (16x32 bf16, row-major source with leading dim lda):
// lane m = lane&15 ; kbase = (lane<16)?0:16 ; VGPR j holds K pair kbase+2j
__device__ inline v16bf load_a_frag(const __bf16* A, int lda, int m0, int k0, int lane) {
  const __bf16* p = A + (size_t)(m0 + (lane & 15)) * lda + k0 + ((lane >> 4) << 4);
  AFrag f;
  f.u[0] = *(const unsigned*)(p + 0);
  f.u[1] = *(const unsigned*)(p + 2);
  f.u[2] = *(const unsigned*)(p + 4);
  f.u[3] = *(const unsigned*)(p + 6);
  f.u[4] = *(const unsigned*)(p + 8);
  f.u[5] = *(const unsigned*)(p + 10);
  f.u[6] = *(const unsigned*)(p + 12);
  f.u[7] = *(const unsigned*)(p + 14);
  return f.v;
}

// ---------------------------------------------------------------------------
// Weight swizzle: fp32 row-major [K][N] -> bf16 B-fragment order.
// Per (nt,kt) 32x16 tile: 32 lanes x 16 elems contiguous.
// lane: n = nt*16 + (lane&15); k = kt*32 + ((lane>=16)?16:0) + e  (sequential K
// per half-wave, the same k order load_a_frag uses).
// ---------------------------------------------------------------------------
__global__ __launch_bounds__(256) void swz_kernel(const float* __restrict__ W,
                                                  __bf16* __restrict__ out,
                                                  int K, int N) {
  int tid = blockIdx.x * 256 + threadIdx.x;
  if (tid >= K * N) return;
  int e    = tid & 15;
  int lane = (tid >> 4) & 31;
  int t    = tid >> 9;
  int KT   = K >> 5;
  int kt   = t % KT;
  int nt   = t / KT;
  int n = nt * 16 + (lane & 15);
  int k = kt * 32 + ((lane >> 4) << 4) + e;
  out[tid] = (__bf16)W[(size_t)k * N + n];
}

// ---------------------------------------------------------------------------
// LayerNorm over C per row; optional fp32 output + bf16 row-major output.
// ---------------------------------------------------------------------------
__global__ __launch_bounds__(256) void ln_kernel(const float* __restrict__ x,
                                                 const float* __restrict__ g,
                                                 const float* __restrict__ bta,
                                                 float* __restrict__ outF,
                                                 __bf16* __restrict__ outH,
                                                 int C) {
  __shared__ float red[256];
  const int row = blockIdx.x;
  const float* xr = x + (size_t)row * C;
  float s = 0.f, q = 0.f;
  for (int i = threadIdx.x; i < C; i += 256) { float v = xr[i]; s += v; q += v * v; }
  red[threadIdx.x] = s; __syncthreads();
  for (int st = 128; st > 0; st >>= 1) {
    if ((int)threadIdx.x < st) red[threadIdx.x] += red[threadIdx.x + st];
    __syncthreads();
  }
  float mean = red[0] / (float)C;
  __syncthreads();
  red[threadIdx.x] = q; __syncthreads();
  for (int st = 128; st > 0; st >>= 1) {
    if ((int)threadIdx.x < st) red[threadIdx.x] += red[threadIdx.x + st];
    __syncthreads();
  }
  float var = red[0] / (float)C - mean * mean;
  float rs = rsqrtf(var + LN_EPS);
  for (int i = threadIdx.x; i < C; i += 256) {
    float v = (xr[i] - mean) * rs * g[i] + bta[i];
    if (outF) outF[(size_t)row * C + i] = v;
    outH[(size_t)row * C + i] = (__bf16)v;
  }
}

// ---------------------------------------------------------------------------
// Generic bf16 WMMA GEMM:  D[M,N] = A[M,K] * Bsw[K,N] + bias  (+ epilogue op)
//   op 0: bias,             bf16 out
//   op 1: bias + exact GELU, bf16 out
//   op 2: bias + addend,    fp32 out
// Block 256 = 8 waves (4 in M x 2 in N); wave tile 32x64; block tile 128x128.
// ---------------------------------------------------------------------------
__global__ __launch_bounds__(256) void gemm_bf16_kernel(
    const __bf16* __restrict__ A, const __bf16* __restrict__ Bsw,
    const float* __restrict__ bias, const float* __restrict__ addend,
    float* __restrict__ outF, __bf16* __restrict__ outH,
    int M, int N, int K, int op) {
  const int lane = threadIdx.x & 31;
  const int wid  = threadIdx.x >> 5;
  const int wy = wid & 3, wx = wid >> 2;
  const int m0 = blockIdx.y * 128 + wy * 32;
  const int n0 = blockIdx.x * 128 + wx * 64;
  const int KT = K >> 5;
  const int ntb = n0 >> 4;
  const v16bf* bp = (const v16bf*)Bsw;

  v8f acc[2][4];
#pragma unroll
  for (int i = 0; i < 2; ++i)
#pragma unroll
    for (int j = 0; j < 4; ++j) acc[i][j] = {};

  for (int kt = 0; kt < KT; ++kt) {
    if (kt + 1 < KT) {
      // gfx1250 global_prefetch_b8 for next K tile (A rows + B fragments)
      __builtin_prefetch(A + (size_t)(m0 + (lane & 15)) * K + (kt + 1) * 32, 0, 1);
      __builtin_prefetch(&bp[(size_t)(ntb * KT + kt + 1) * 32 + lane], 0, 1);
    }
    v16bf a0 = load_a_frag(A, K, m0,      kt * 32, lane);
    v16bf a1 = load_a_frag(A, K, m0 + 16, kt * 32, lane);
#pragma unroll
    for (int j = 0; j < 4; ++j) {
      v16bf b = bp[(size_t)((ntb + j) * KT + kt) * 32 + lane];
      acc[0][j] = wmma_bf16(a0, b, acc[0][j]);
      acc[1][j] = wmma_bf16(a1, b, acc[1][j]);
    }
  }

#pragma unroll
  for (int i = 0; i < 2; ++i)
#pragma unroll
    for (int j = 0; j < 4; ++j)
#pragma unroll
      for (int r = 0; r < 8; ++r) {
        int m = m0 + i * 16 + r + ((lane >> 4) << 3);
        int n = n0 + j * 16 + (lane & 15);
        size_t idx = (size_t)m * N + n;
        float v = acc[i][j][r] + bias[n];
        if (op == 1) v = 0.5f * v * (1.0f + erff(v * 0.70710678118654752f));
        if (op == 2) outF[idx] = v + addend[idx];
        else         outH[idx] = (__bf16)v;
      }
}

// ---------------------------------------------------------------------------
// Causal flash attention. Reference computes scores[i,j] = k_i . q_j, so the
// "query" tensor here is the k-projection and the "key" tensor is q.
// One wave per 16 query rows; KV tiles of 32; per-wave LDS staging with
// gfx1250 async global->LDS copies for Q/K tiles.
// Epilogue fuses residual: x2 = xln1 + attn_out.
// ---------------------------------------------------------------------------
#define AT_WAVES 4
__global__ __launch_bounds__(128) void attn_kernel(
    const __bf16* __restrict__ qb, const __bf16* __restrict__ kb,
    const __bf16* __restrict__ vb, const float* __restrict__ xln1,
    float* __restrict__ x2) {
  __shared__ __attribute__((aligned(32))) __bf16 Qs[AT_WAVES][16 * 64];
  __shared__ __attribute__((aligned(32))) __bf16 Ks[AT_WAVES][32 * 64];
  __shared__ __attribute__((aligned(32))) __bf16 Vt[AT_WAVES][64 * 32];  // transposed [d][j]
  __shared__ __attribute__((aligned(32))) __bf16 Ps[AT_WAVES][16 * 32];

  const int T = 2048, C = 768, H = 12, QT = T / 16;
  const int lane = threadIdx.x & 31;
  const int w    = threadIdx.x >> 5;
  int gw  = blockIdx.x * AT_WAVES + w;
  int qt  = gw % QT;
  int rem = gw / QT;
  int h   = rem % H;
  int b   = rem / H;

  const __bf16* Qg = kb + ((size_t)b * T) * C + h * 64;  // queries  = k tensor
  const __bf16* Kg = qb + ((size_t)b * T) * C + h * 64;  // keys     = q tensor
  const __bf16* Vg = vb + ((size_t)b * T) * C + h * 64;

  const unsigned qs_base = lds_off_of(&Qs[w][0]);
  const unsigned ks_base = lds_off_of(&Ks[w][0]);

  // stage Q tile (16x64) via async copies
#pragma unroll
  for (int it = 0; it < 4; ++it) {
    int v = lane + it * 32;           // 128 chunks of 8 bf16
    int row = v >> 3, c8 = (v & 7) * 8;
    async_copy_b128(qs_base + (unsigned)(row * 64 + c8) * 2,
                    &Qg[(size_t)(qt * 16 + row) * C + c8]);
  }
  asm volatile("s_wait_asynccnt 0" ::: "memory");
  v16bf aq0 = load_a_frag(Qs[w], 64, 0, 0,  lane);
  v16bf aq1 = load_a_frag(Qs[w], 64, 0, 32, lane);

  v8f O[4];
#pragma unroll
  for (int dt = 0; dt < 4; ++dt) O[dt] = {};
  float Mr[8], Lr[8];
#pragma unroll
  for (int r = 0; r < 8; ++r) { Mr[r] = -__builtin_inff(); Lr[r] = 0.f; }

  const int n_j   = lane & 15;
  const int kb16  = (lane >> 4) << 4;
  const int mrow0 = qt * 16 + ((lane >> 4) << 3);
  const int jt_max = (qt * 16 + 15) >> 5;

  for (int jt = 0; jt <= jt_max; ++jt) {
    // DS and ASYNC queues are mutually unordered: make sure last iteration's
    // LDS reads retired before async stores may overwrite the K tile.
    asm volatile("s_wait_dscnt 0" ::: "memory");
    // stage K tile (32x64 row-major) via async copies
#pragma unroll
    for (int it = 0; it < 8; ++it) {
      int v = lane + it * 32;         // 256 chunks of 8 bf16
      int row = v >> 3, c8 = (v & 7) * 8;
      async_copy_b128(ks_base + (unsigned)(row * 64 + c8) * 2,
                      &Kg[(size_t)(jt * 32 + row) * C + c8]);
    }
    // stage V tile transposed: Vt[d][j], lane = row j (register transpose)
    {
      const __bf16* vr = &Vg[(size_t)(jt * 32 + lane) * C];
#pragma unroll
      for (int c = 0; c < 64; c += 8) {
        uint4 t = *(const uint4*)&vr[c];
        const __bf16* tb = (const __bf16*)&t;
#pragma unroll
        for (int e = 0; e < 8; ++e) Vt[w][(c + e) * 32 + lane] = tb[e];
      }
    }
    asm volatile("s_wait_asynccnt 0" ::: "memory");
    asm volatile("s_wait_dscnt 0" ::: "memory");

    // scores: S[m, j] over 32 kv cols (two 16-wide n-tiles), contraction d=64
    v8f s0 = {}, s1 = {};
    s0 = wmma_bf16(aq0, *(const v16bf*)&Ks[w][n_j * 64 + 0  + kb16], s0);
    s0 = wmma_bf16(aq1, *(const v16bf*)&Ks[w][n_j * 64 + 32 + kb16], s0);
    s1 = wmma_bf16(aq0, *(const v16bf*)&Ks[w][(16 + n_j) * 64 + 0  + kb16], s1);
    s1 = wmma_bf16(aq1, *(const v16bf*)&Ks[w][(16 + n_j) * 64 + 32 + kb16], s1);

    // online softmax (row stats reduced across the 16-lane half-wave)
    float al[8];
    int j0 = jt * 32 + n_j, j1 = j0 + 16;
#pragma unroll
    for (int r = 0; r < 8; ++r) {
      int i = mrow0 + r;
      float v0 = s0[r] * 0.125f; if (j0 > i) v0 = -__builtin_inff();
      float v1 = s1[r] * 0.125f; if (j1 > i) v1 = -__builtin_inff();
      float mx = fmaxf(v0, v1);
#pragma unroll
      for (int off = 1; off < 16; off <<= 1) mx = fmaxf(mx, __shfl_xor(mx, off, 32));
      float nM = fmaxf(Mr[r], mx);
      float a  = __expf(Mr[r] - nM);
      float p0 = __expf(v0 - nM);
      float p1 = __expf(v1 - nM);
      float rs = p0 + p1;
#pragma unroll
      for (int off = 1; off < 16; off <<= 1) rs += __shfl_xor(rs, off, 32);
      Lr[r] = Lr[r] * a + rs;
      Mr[r] = nM;
      al[r] = a;
      s0[r] = p0;
      s1[r] = p1;
    }
#pragma unroll
    for (int dt = 0; dt < 4; ++dt)
#pragma unroll
      for (int r = 0; r < 8; ++r) O[dt][r] *= al[r];

    // P (D-layout) -> LDS -> A-layout fragment
#pragma unroll
    for (int r = 0; r < 8; ++r) {
      int m = r + ((lane >> 4) << 3);
      Ps[w][m * 32 + n_j]      = (__bf16)s0[r];
      Ps[w][m * 32 + 16 + n_j] = (__bf16)s1[r];
    }
    asm volatile("s_wait_dscnt 0" ::: "memory");
    v16bf ap = load_a_frag(Ps[w], 32, 0, 0, lane);

    // O += P @ V  (B fragments contiguous from transposed Vt)
#pragma unroll
    for (int dt = 0; dt < 4; ++dt) {
      v16bf bv = *(const v16bf*)&Vt[w][(dt * 16 + n_j) * 32 + kb16];
      O[dt] = wmma_bf16(ap, bv, O[dt]);
    }
    asm volatile("" ::: "memory");  // keep next tile's LDS writes after these reads
  }

  // normalize + fused residual: x2 = xln1 + y
#pragma unroll
  for (int dt = 0; dt < 4; ++dt)
#pragma unroll
    for (int r = 0; r < 8; ++r) {
      int row = qt * 16 + r + ((lane >> 4) << 3);
      int col = h * 64 + dt * 16 + n_j;
      size_t idx = ((size_t)b * T + row) * C + col;
      x2[idx] = xln1[idx] + O[dt][r] / Lr[r];
    }
}

// ---------------------------------------------------------------------------
extern "C" void kernel_launch(void* const* d_in, const int* in_sizes, int n_in,
                              void* d_out, int out_size, void* d_ws, size_t ws_size,
                              hipStream_t stream) {
  (void)in_sizes; (void)n_in; (void)out_size; (void)ws_size;
  const float* x    = (const float*)d_in[0];
  const float* ln1g = (const float*)d_in[1];
  const float* ln1b = (const float*)d_in[2];
  const float* Wq = (const float*)d_in[3];  const float* bq = (const float*)d_in[4];
  const float* Wk = (const float*)d_in[5];  const float* bk = (const float*)d_in[6];
  const float* Wv = (const float*)d_in[7];  const float* bv = (const float*)d_in[8];
  const float* ln2g = (const float*)d_in[9];  const float* ln2b = (const float*)d_in[10];
  const float* W1 = (const float*)d_in[11]; const float* b1 = (const float*)d_in[12];
  const float* W2 = (const float*)d_in[13]; const float* b2 = (const float*)d_in[14];
  float* out = (float*)d_out;

  const int Mrows = 4096, C = 768, F = 3072;

  char* ws = (char*)d_ws;
  auto alloc = [&](size_t bytes) -> char* {
    char* p = ws;
    ws += (bytes + 255) & ~(size_t)255;
    return p;
  };
  float*  xln1F = (float*)  alloc((size_t)Mrows * C * 4);
  __bf16* xln1H = (__bf16*) alloc((size_t)Mrows * C * 2);
  __bf16* wqs   = (__bf16*) alloc((size_t)C * C * 2);
  __bf16* wks   = (__bf16*) alloc((size_t)C * C * 2);
  __bf16* wvs   = (__bf16*) alloc((size_t)C * C * 2);
  __bf16* w1s   = (__bf16*) alloc((size_t)C * F * 2);
  __bf16* w2s   = (__bf16*) alloc((size_t)F * C * 2);
  __bf16* qH    = (__bf16*) alloc((size_t)Mrows * C * 2);
  __bf16* kH    = (__bf16*) alloc((size_t)Mrows * C * 2);
  __bf16* vH    = (__bf16*) alloc((size_t)Mrows * C * 2);
  float*  x2F   = (float*)  alloc((size_t)Mrows * C * 4);
  __bf16* hH    = (__bf16*) alloc((size_t)Mrows * C * 2);
  __bf16* g1H   = (__bf16*) alloc((size_t)Mrows * F * 2);

  // 1) weight swizzle fp32 -> bf16 fragment order
  swz_kernel<<<(C * C + 255) / 256, 256, 0, stream>>>(Wq, wqs, C, C);
  swz_kernel<<<(C * C + 255) / 256, 256, 0, stream>>>(Wk, wks, C, C);
  swz_kernel<<<(C * C + 255) / 256, 256, 0, stream>>>(Wv, wvs, C, C);
  swz_kernel<<<(C * F + 255) / 256, 256, 0, stream>>>(W1, w1s, C, F);
  swz_kernel<<<(F * C + 255) / 256, 256, 0, stream>>>(W2, w2s, F, C);

  // 2) LN1 -> fp32 (residual) + bf16 (GEMM A)
  ln_kernel<<<Mrows, 256, 0, stream>>>(x, ln1g, ln1b, xln1F, xln1H, C);

  // 3) QKV projections
  gemm_bf16_kernel<<<dim3(C / 128, Mrows / 128), 256, 0, stream>>>(
      xln1H, wqs, bq, nullptr, nullptr, qH, Mrows, C, C, 0);
  gemm_bf16_kernel<<<dim3(C / 128, Mrows / 128), 256, 0, stream>>>(
      xln1H, wks, bk, nullptr, nullptr, kH, Mrows, C, C, 0);
  gemm_bf16_kernel<<<dim3(C / 128, Mrows / 128), 256, 0, stream>>>(
      xln1H, wvs, bv, nullptr, nullptr, vH, Mrows, C, C, 0);

  // 4) causal flash attention + residual (x2 = xln1 + y)
  attn_kernel<<<(2 * 12 * (2048 / 16)) / AT_WAVES, 32 * AT_WAVES, 0, stream>>>(
      qH, kH, vH, xln1F, x2F);

  // 5) LN2 -> bf16
  ln_kernel<<<Mrows, 256, 0, stream>>>(x2F, ln2g, ln2b, nullptr, hH, C);

  // 6) MLP up + exact GELU
  gemm_bf16_kernel<<<dim3(F / 128, Mrows / 128), 256, 0, stream>>>(
      hH, w1s, b1, nullptr, nullptr, g1H, Mrows, F, C, 1);

  // 7) MLP down + bias + residual -> fp32 output
  gemm_bf16_kernel<<<dim3(C / 128, Mrows / 128), 256, 0, stream>>>(
      g1H, w2s, b2, x2F, out, nullptr, Mrows, C, F, 2);
}
